// GIN_3624952397847
// MI455X (gfx1250) — compile-verified, matrix operand order checked by reading, not run on hardware
//
#include <hip/hip_runtime.h>
#include <hip/hip_bf16.h>
#include <stdint.h>

typedef _Float16 v16h __attribute__((ext_vector_type(16)));
typedef _Float16 v8h  __attribute__((ext_vector_type(8)));
typedef float    v8f  __attribute__((ext_vector_type(8)));

#define D    128
#define BSTR 136   // LDS row stride in halves: 272B = 16B-aligned, breaks bank conflicts

__device__ __forceinline__ float lrelu(float x){ return x > 0.f ? x : 0.01f * x; }

// ---------------------------------------------------------------- utility
__global__ void zero_kernel(float* p, int n){
    int i = blockIdx.x * blockDim.x + threadIdx.x;
    if (i < n) p[i] = 0.f;
}

__global__ void one_hot_kernel(const int* __restrict__ deg, float* __restrict__ h, int N){
    // one thread per node, writes a full 128-float row as 32x float4
    int node = blockIdx.x * blockDim.x + threadIdx.x;
    if (node >= N) return;
    int d = deg[node];
    float4* row = (float4*)(h + (size_t)node * D);
    #pragma unroll
    for (int q = 0; q < D / 4; ++q) row[q] = make_float4(0.f, 0.f, 0.f, 0.f);
    h[(size_t)node * D + d] = 1.f;
}

__global__ void init_agg_kernel(const float* __restrict__ h, const float* __restrict__ eps,
                                int l, float* __restrict__ agg, int total4){
    int i = blockIdx.x * blockDim.x + threadIdx.x;   // float4 index
    if (i >= total4) return;
    float s = 1.f + eps[l];
    float4 v = ((const float4*)h)[i];
    v.x *= s; v.y *= s; v.z *= s; v.w *= s;
    ((float4*)agg)[i] = v;
}

// one wave per edge: 32 lanes x float4 = 128 features, coalesced gather + atomic scatter
__global__ void scatter_kernel(const int* __restrict__ src, const int* __restrict__ dst,
                               const float* __restrict__ h, float* __restrict__ agg,
                               long long E){
    long long idx = (long long)blockIdx.x * blockDim.x + threadIdx.x;
    long long e = idx >> 5;
    if (e >= E) return;
    int c = (int)(idx & 31) << 2;
    long long s = src[e], d = dst[e];
    const float4 v = *(const float4*)(h + s * D + c);
    float* a = agg + d * D + c;
    atomicAdd(a + 0, v.x); atomicAdd(a + 1, v.y);
    atomicAdd(a + 2, v.z); atomicAdd(a + 3, v.w);
}

// ---------------------------------------------------------------- WMMA GEMM
// C[N,128] = act(A[N,128] @ W[128,128] + bias), optional per-column sum/sumsq atomics.
// Block: 256 thr = 8 waves -> 128x128 output tile. Wave w: rows w*16..w*16+15, all 8 col tiles.
template<bool AHALF, bool OUTHALF, bool RELU, bool STATS>
__global__ void gemm_kernel(const void* __restrict__ A, const float* __restrict__ W,
                            const float* __restrict__ bias, void* __restrict__ Cout,
                            int N, float* __restrict__ sums, float* __restrict__ sumsq){
    __shared__ _Float16 Bs[D * BSTR];          // W transposed: Bs[n*BSTR + k]
    const int tid = threadIdx.x;
    for (int i = tid; i < D * D; i += 256){
        int k = i >> 7, n = i & (D - 1);
        Bs[n * BSTR + k] = (_Float16)W[i];     // W[k*128+n], coalesced read
    }

    const int lane = tid & 31, wid = tid >> 5;
    const int hlf  = lane >> 4;                // K-half select per ISA A layout
    const int mrow = blockIdx.x * 128 + wid * 16 + (lane & 15);
    // clamp instead of guard: C row m depends only on A row m; rows >= N are
    // never stored and never counted, so garbage there is harmless.
    const int arow = mrow < N ? mrow : N - 1;

    // Hoist A fragments for all 4 K-blocks (fp32->f16 convert or raw f16 16B loads)
    v16h afr[4];
    #pragma unroll
    for (int kb = 0; kb < 4; ++kb){
        v16h a;
        const int k0 = kb * 32 + hlf * 8;
        if constexpr (AHALF){
            const _Float16* Ah = (const _Float16*)A + (size_t)arow * D + k0;
            v8h lo = *(const v8h*)Ah;
            v8h hi = *(const v8h*)(Ah + 16);
            #pragma unroll
            for (int j = 0; j < 8; ++j){ a[j] = lo[j]; a[8 + j] = hi[j]; }
        } else {
            const float* Af = (const float*)A + (size_t)arow * D + k0;
            #pragma unroll
            for (int j = 0; j < 8; ++j){ a[j] = (_Float16)Af[j]; a[8 + j] = (_Float16)Af[16 + j]; }
        }
        afr[kb] = a;
    }
    __syncthreads();

    const int  rowbase = blockIdx.x * 128 + wid * 16;
    const bool full    = (blockIdx.x * 128 + 128) <= N;   // wave-uniform
    const int  nl      = lane & 15;
    #pragma unroll
    for (int nt = 0; nt < 8; ++nt){
        const int ncol = nt * 16 + nl;
        v8f c = {};
        #pragma unroll
        for (int kb = 0; kb < 4; ++kb){
            const _Float16* bp = Bs + ncol * BSTR + kb * 32 + hlf * 8;
            v8h blo = *(const v8h*)bp;
            v8h bhi = *(const v8h*)(bp + 16);
            v16h b;
            #pragma unroll
            for (int j = 0; j < 8; ++j){ b[j] = blo[j]; b[8 + j] = bhi[j]; }
            c = __builtin_amdgcn_wmma_f32_16x16x32_f16(false, afr[kb], false, b,
                                                       (short)0, c, false, false);
        }
        const float bia = bias[ncol];
        float s = 0.f, s2 = 0.f;
        if (full){
            // fast path: no per-row guards, straightline epilogue
            #pragma unroll
            for (int v = 0; v < 8; ++v){
                const int row = rowbase + v + 8 * hlf;   // C/D layout: VGPR v -> row v + 8*half
                float r = c[v] + bia;
                if constexpr (RELU) r = lrelu(r);
                if constexpr (OUTHALF)
                    ((_Float16*)Cout)[(size_t)row * D + ncol] = (_Float16)r;
                else
                    ((float*)Cout)[(size_t)row * D + ncol] = r;
                if constexpr (STATS){ s += r; s2 += r * r; }
            }
        } else {
            #pragma unroll
            for (int v = 0; v < 8; ++v){
                const int row = rowbase + v + 8 * hlf;
                if (row < N){
                    float r = c[v] + bia;
                    if constexpr (RELU) r = lrelu(r);
                    if constexpr (OUTHALF)
                        ((_Float16*)Cout)[(size_t)row * D + ncol] = (_Float16)r;
                    else
                        ((float*)Cout)[(size_t)row * D + ncol] = r;
                    if constexpr (STATS){ s += r; s2 += r * r; }
                }
            }
        }
        if constexpr (STATS){
            atomicAdd(&sums[ncol],  s);
            atomicAdd(&sumsq[ncol], s2);
        }
    }
}

// ---------------------------------------------------------------- BatchNorm
__global__ void bn_stats_kernel(const float* __restrict__ sums, const float* __restrict__ sumsq,
                                const float* __restrict__ g, const float* __restrict__ b,
                                float invN, float* __restrict__ scale, float* __restrict__ shift){
    int c = threadIdx.x;
    float mean = sums[c] * invN;
    float var  = sumsq[c] * invN - mean * mean;
    float sc   = g[c] * rsqrtf(var + 1e-5f);
    scale[c] = sc;
    shift[c] = b[c] - mean * sc;
}

__global__ void bn_apply_kernel(const float* __restrict__ t2, const float* __restrict__ scale,
                                const float* __restrict__ shift, float* __restrict__ h, int total4){
    int i = blockIdx.x * blockDim.x + threadIdx.x;   // float4 index
    if (i >= total4) return;
    int c = (i << 2) & (D - 1);
    float4 v = ((const float4*)t2)[i];
    v.x = lrelu(v.x * scale[c + 0] + shift[c + 0]);
    v.y = lrelu(v.y * scale[c + 1] + shift[c + 1]);
    v.z = lrelu(v.z * scale[c + 2] + shift[c + 2]);
    v.w = lrelu(v.w * scale[c + 3] + shift[c + 3]);
    ((float4*)h)[i] = v;
}

// ---------------------------------------------------------------- pooling (batch_vec sorted)
__global__ void pool_kernel(const float* __restrict__ h, const int* __restrict__ batch,
                            float* __restrict__ pooled, int layer, int N, int ldz){
    int c = threadIdx.x;               // feature column 0..127
    int base = blockIdx.x * 32;
    if (base >= N) return;
    int end = base + 32; if (end > N) end = N;
    float acc = 0.f;
    int pb = batch[base];
    for (int i = base; i < end; ++i){
        int b = batch[i];
        if (b != pb){
            atomicAdd(&pooled[(size_t)pb * ldz + layer * D + c], acc);
            acc = 0.f; pb = b;
        }
        acc += h[(size_t)i * D + c];
    }
    atomicAdd(&pooled[(size_t)pb * ldz + layer * D + c], acc);
}

// ---------------------------------------------------------------- classifier head (tiny)
__global__ void head_kernel(const float* __restrict__ pooled,
                            const float* __restrict__ Wc1, const float* __restrict__ bc1,
                            const float* __restrict__ Wc2, const float* __restrict__ bc2,
                            float* __restrict__ out, int cls_in, int ch, int nc){
    __shared__ float pl[1024];
    __shared__ float hid[512];
    int g = blockIdx.x, t = threadIdx.x;
    for (int k = t; k < cls_in; k += blockDim.x)
        pl[k] = lrelu(pooled[(size_t)g * cls_in + k]);
    __syncthreads();
    float acc = bc1[t];
    for (int k = 0; k < cls_in; ++k) acc += pl[k] * Wc1[(size_t)k * ch + t];
    hid[t] = lrelu(acc);
    __syncthreads();
    if (t < nc){
        float o = bc2[t];
        for (int k = 0; k < ch; ++k) o += hid[k] * Wc2[(size_t)k * nc + t];
        out[(size_t)g * nc + t] = o;
    }
}

// ---------------------------------------------------------------- launch
extern "C" void kernel_launch(void* const* d_in, const int* in_sizes, int n_in,
                              void* d_out, int out_size, void* d_ws, size_t ws_size,
                              hipStream_t stream){
    const int*   node_deg = (const int*)d_in[0];
    const int*   edge_ix  = (const int*)d_in[1];
    const int*   batch    = (const int*)d_in[2];
    const float* eps  = (const float*)d_in[3];
    const float* W1   = (const float*)d_in[4];
    const float* b1   = (const float*)d_in[5];
    const float* W2   = (const float*)d_in[6];
    const float* b2   = (const float*)d_in[7];
    const float* bn_g = (const float*)d_in[8];
    const float* bn_b = (const float*)d_in[9];
    const float* Wc1  = (const float*)d_in[10];
    const float* bc1  = (const float*)d_in[11];
    const float* Wc2  = (const float*)d_in[12];
    const float* bc2  = (const float*)d_in[13];

    const int N       = in_sizes[0];
    const long long E = (long long)in_sizes[1] / 2;
    const int L       = in_sizes[3];
    const int CH      = in_sizes[11];        // 256
    const int NC      = in_sizes[13];        // 10
    const int G       = out_size / NC;       // 64
    const int ldz     = (L + 1) * D;         // 640

    uint8_t* ws = (uint8_t*)d_ws;
    size_t off = 0;
    auto carve = [&](size_t bytes) -> void* {
        void* p = ws + off;
        off += (bytes + 255) & ~(size_t)255;
        return p;
    };
    float*    agg    = (float*)carve((size_t)N * D * sizeof(float));    // reused as GEMM2 out
    float*    h      = (float*)carve((size_t)N * D * sizeof(float));
    _Float16* t1     = (_Float16*)carve((size_t)N * D * sizeof(_Float16));
    float*    pooled = (float*)carve((size_t)G * ldz * sizeof(float));
    float*    stats  = (float*)carve(2 * D * sizeof(float));            // sums | sumsq
    float*    scale  = (float*)carve(D * sizeof(float));
    float*    shift  = (float*)carve(D * sizeof(float));
    (void)ws_size; (void)n_in;

    const int TB  = 256;
    const int ND4 = N * D / 4;

    zero_kernel<<<(G * ldz + TB - 1) / TB, TB, 0, stream>>>(pooled, G * ldz);
    one_hot_kernel<<<(N + TB - 1) / TB, TB, 0, stream>>>(node_deg, h, N);
    pool_kernel<<<(N + 31) / 32, D, 0, stream>>>(h, batch, pooled, 0, N, ldz);

    for (int l = 0; l < L; ++l){
        init_agg_kernel<<<(ND4 + TB - 1) / TB, TB, 0, stream>>>(h, eps, l, agg, ND4);
        long long chunks = E * 32;
        scatter_kernel<<<(int)((chunks + TB - 1) / TB), TB, 0, stream>>>(
            edge_ix, edge_ix + E, h, agg, E);
        gemm_kernel<false, true, true, false><<<(N + 127) / 128, TB, 0, stream>>>(
            agg, W1 + (size_t)l * D * D, b1 + l * D, t1, N, nullptr, nullptr);
        zero_kernel<<<1, 2 * D, 0, stream>>>(stats, 2 * D);
        gemm_kernel<true, false, false, true><<<(N + 127) / 128, TB, 0, stream>>>(
            t1, W2 + (size_t)l * D * D, b2 + l * D, agg, N, stats, stats + D);
        bn_stats_kernel<<<1, D, 0, stream>>>(stats, stats + D, bn_g + l * D, bn_b + l * D,
                                             1.0f / (float)N, scale, shift);
        bn_apply_kernel<<<(ND4 + TB - 1) / TB, TB, 0, stream>>>(agg, scale, shift, h, ND4);
        pool_kernel<<<(N + 31) / 32, D, 0, stream>>>(h, batch, pooled, l + 1, N, ldz);
    }

    head_kernel<<<G, CH, 0, stream>>>(pooled, Wc1, bc1, Wc2, bc2, (float*)d_out, ldz, CH, NC);
}
